// DTransformer_10213432230089
// MI455X (gfx1250) — compile-verified, matrix operand order checked by reading, not run on hardware
//
#include <hip/hip_runtime.h>
#include <hip/hip_bf16.h>

// ---------------------------------------------------------------------------
// GPT forward pass for MI455X (gfx1250).
//  - All GEMMs via v_wmma_f32_16x16x32_f16 (f16 in, f32 accumulate)
//  - Global->LDS tile staging via TDM tensor_load_to_lds (double-buffered),
//    with TDM LDS padding giving an 80B row stride (bank-conflict free)
//  - Weights pre-transposed to W^T[N][K] at f32->f16 conversion, so both
//    A and B tiles are row-major with contiguous K
//  - Flash attention: 1 wave / 16-query tile, online softmax, WMMA for
//    Q@K^T and P@V, P restaged through LDS (C-layout -> A-layout)
// ---------------------------------------------------------------------------

typedef _Float16 f16;
typedef __attribute__((ext_vector_type(16))) _Float16 v16h;
typedef __attribute__((ext_vector_type(8)))  _Float16 v8h;
typedef __attribute__((ext_vector_type(8)))  float    v8f;
typedef __attribute__((ext_vector_type(4)))  unsigned int v4u;
typedef __attribute__((ext_vector_type(8)))  int      v8i;
typedef __attribute__((ext_vector_type(4)))  int      v4i;

#define CB   2
#define CT   2048
#define CD   1024
#define CH   16
#define CHS  64
#define CL   4
#define CV   32000
#define CBT  (CB*CT)
#define CFD  (4*CD)

#define GF_BIAS 1
#define GF_RELU 2
#define GF_RES  4
#define GF_F32  8
#define GF_F16  16

#if defined(__HIP_DEVICE_COMPILE__) && __has_builtin(__builtin_amdgcn_tensor_load_to_lds) && __has_builtin(__builtin_amdgcn_s_wait_tensorcnt)
#define GPT_TDM 1
#else
#define GPT_TDM 0
#endif

// -------------------- helpers --------------------

// 16-element WMMA operand fragment: two contiguous 8xf16 (16B) loads.
// p points at local-K offset kb (lane<16 -> 0, lane>=16 -> 8); 2nd load at +16.
__device__ __forceinline__ v16h ldfrag(const f16* p) {
    v8h lo = *(const v8h*)p;
    v8h hi = *(const v8h*)(p + 16);
    v16h r;
#pragma unroll
    for (int e = 0; e < 8; ++e) { r[e] = lo[e]; r[e + 8] = hi[e]; }
    return r;
}

__device__ __forceinline__ float rmax16(float v) {
#pragma unroll
    for (int m = 8; m >= 1; m >>= 1) v = fmaxf(v, __shfl_xor(v, m, 16));
    return v;
}
__device__ __forceinline__ float rsum16(float v) {
#pragma unroll
    for (int m = 8; m >= 1; m >>= 1) v += __shfl_xor(v, m, 16);
    return v;
}

#if GPT_TDM
// Issue one TDM 2D tile load: tile_dim = 32 (K, contiguous) x 128 (rows),
// element = 2B f16, LDS pad: +4 DWORDs after every 16 DWORDs -> row stride
// 32 f16 + 8 f16 = 40 f16 = 80B (16B aligned, conflict-free ds_load_b128).
// D# packing per CDNA5 ISA.
// Toolchain builtin is the 6-arg form:
//   (uint32x4 g0, int32x8 g1, int32x4 g2, int32x4 g3, int32x8 gx, i32 cpol)
__device__ __forceinline__ void tdm_load_tile(const f16* gptr, unsigned lds_off,
                                              int dim0, int dim1, int stride0) {
    unsigned long long ga = (unsigned long long)(uintptr_t)gptr;
    v4u g0;
    g0[0] = 1u;                                        // count=1, user desc
    g0[1] = lds_off;                                   // lds_addr (bytes)
    g0[2] = (unsigned)(ga & 0xFFFFFFFFu);              // global_addr[31:0]
    g0[3] = (unsigned)((ga >> 32) & 0x1FFFFFFu)        // global_addr[56:32]
          | (2u << 30);                                // type=2 ("image")
    v8i g1;
    g1[0] = (int)((1u << 16)                           // data_size=1 -> 2B
                | (1u << 20)                           // pad_enable
                | (3u << 22)                           // pad_interval: 16 DW
                | (3u << 25));                         // pad_amount:  4 DW
    g1[1] = (int)(((unsigned)dim0 & 0xFFFFu) << 16);   // tensor_dim0[15:0]
    g1[2] = (int)(((unsigned)dim0 >> 16)               // tensor_dim0[31:16]
                | (((unsigned)dim1 & 0xFFFFu) << 16)); // tensor_dim1[15:0]
    g1[3] = (int)(((unsigned)dim1 >> 16)               // tensor_dim1[31:16]
                | (32u << 16));                        // tile_dim0 = 32
    g1[4] = 128;                                       // tile_dim1 = 128
    g1[5] = stride0;                                   // tensor_dim0_stride
    g1[6] = 0;
    g1[7] = 0;
    v4i z4 = {0, 0, 0, 0};
    v8i z8 = {0, 0, 0, 0, 0, 0, 0, 0};
    __builtin_amdgcn_tensor_load_to_lds(g0, g1, z4, z4, z8, 0);
}
#endif

// -------------------- misc kernels --------------------

__global__ __launch_bounds__(256) void k_embed(const int* __restrict__ x,
                                               const float* __restrict__ emb,
                                               const float* __restrict__ pos,
                                               float* __restrict__ h) {
    int row = blockIdx.x;
    int tok = x[row];
    int tp  = row % CT;
    const float* er = emb + (size_t)tok * CD;
    const float* pr = pos + (size_t)tp * CD;
    float* hr = h + (size_t)row * CD;
#pragma unroll
    for (int k = 0; k < 4; ++k) {
        int i = threadIdx.x + k * 256;
        hr[i] = er[i] + pr[i];
    }
}

// fp32 [Kd][Nd] -> f16 transposed [Nd][Kd], 32x32 LDS tiles.
__global__ __launch_bounds__(256) void k_cvt_tr(const float* __restrict__ in,
                                                f16* __restrict__ out,
                                                int Kd, int Nd) {
    __shared__ f16 tile[32][33];
    int n0 = blockIdx.x * 32;
    int k0 = blockIdx.y * 32;
    int tx = threadIdx.x & 31;
    int ty = threadIdx.x >> 5;          // 0..7
#pragma unroll
    for (int r = ty; r < 32; r += 8)
        tile[r][tx] = (f16)in[(size_t)(k0 + r) * Nd + n0 + tx];
    __syncthreads();
#pragma unroll
    for (int r = ty; r < 32; r += 8)
        out[(size_t)(n0 + r) * Kd + k0 + tx] = tile[tx][r];
}

// LayerNorm over D=1024, fp32 in -> f16 out. One block (256 thr) per token.
__global__ __launch_bounds__(256) void k_layernorm(const float* __restrict__ x,
                                                   const float* __restrict__ g,
                                                   const float* __restrict__ b,
                                                   f16* __restrict__ out) {
    __shared__ float red[256];
    int row = blockIdx.x;
    int t = threadIdx.x;
    const float* xr = x + (size_t)row * CD;
    float v[4], s = 0.f;
#pragma unroll
    for (int k = 0; k < 4; ++k) { v[k] = xr[t + k * 256]; s += v[k]; }
    red[t] = s; __syncthreads();
    for (int o = 128; o > 0; o >>= 1) { if (t < o) red[t] += red[t + o]; __syncthreads(); }
    float mean = red[0] * (1.f / CD);
    __syncthreads();
    float s2 = 0.f;
#pragma unroll
    for (int k = 0; k < 4; ++k) { float d = v[k] - mean; s2 += d * d; }
    red[t] = s2; __syncthreads();
    for (int o = 128; o > 0; o >>= 1) { if (t < o) red[t] += red[t + o]; __syncthreads(); }
    float rstd = rsqrtf(red[0] * (1.f / CD) + 1e-5f);
    f16* orow = out + (size_t)row * CD;
#pragma unroll
    for (int k = 0; k < 4; ++k) {
        int i = t + k * 256;
        orow[i] = (f16)((v[k] - mean) * rstd * g[i] + b[i]);
    }
}

// -------------------- WMMA GEMM --------------------
// C[M,N](f32) = A[M,K](f16) @ Bt[N,K](f16)^T  (+bias)(+relu)(+residual).
// Block tile 128x128, 8 waves, wave tile 64x32 (4x2 wmma). TDM double-buffer.
__global__ __launch_bounds__(256) void k_gemm_wmma(
        const f16* __restrict__ A, const f16* __restrict__ Bt,
        const float* __restrict__ bias, const float* __restrict__ resid,
        float* __restrict__ outF, f16* __restrict__ outH,
        int M, int N, int K, int flags) {
    __shared__ f16 sA[2][128 * 40];   // [row][k], stride 40 f16 (80B)
    __shared__ f16 sB[2][128 * 40];   // [n][k]

    const int t    = threadIdx.x;
    const int lane = t & 31;
    const int w    = t >> 5;
    const int wm   = (w & 1) * 64;
    const int wn   = (w >> 1) * 32;
    const int bm   = blockIdx.y * 128;
    const int bn   = blockIdx.x * 128;
    const int half = lane >> 4;
    const int mr   = lane & 15;
    const int kb   = half * 8;
    const int nk   = K >> 5;

    v8f acc[4][2];
#pragma unroll
    for (int i = 0; i < 4; ++i)
#pragma unroll
        for (int j = 0; j < 2; ++j)
#pragma unroll
            for (int r = 0; r < 8; ++r) acc[i][j][r] = 0.f;

#if GPT_TDM
    const bool isw0 = (w == 0);
    const f16* Abase = A  + (size_t)bm * K;
    const f16* Bbase = Bt + (size_t)bn * K;
    if (isw0) {
        tdm_load_tile(Abase, (unsigned)(uintptr_t)&sA[0][0], K, M, K);
        tdm_load_tile(Bbase, (unsigned)(uintptr_t)&sB[0][0], K, N, K);
    }
    for (int i = 0; i < nk; ++i) {
        if (isw0) {
            if (i + 1 < nk) {
                int nb = (i + 1) & 1;
                tdm_load_tile(Abase + (i + 1) * 32, (unsigned)(uintptr_t)&sA[nb][0], K, M, K);
                tdm_load_tile(Bbase + (i + 1) * 32, (unsigned)(uintptr_t)&sB[nb][0], K, N, K);
                __builtin_amdgcn_s_wait_tensorcnt((short)2);  // tile i landed
            } else {
                __builtin_amdgcn_s_wait_tensorcnt((short)0);
            }
        }
        __syncthreads();
        const f16* pa = &sA[i & 1][0];
        const f16* pb = &sB[i & 1][0];
#else
    for (int i = 0; i < nk; ++i) {
        int k0 = i << 5;
        // synchronous staging fallback (both tiles row-major, contiguous K)
#pragma unroll
        for (int l = 0; l < 2; ++l) {
            int e   = t + l * 256;          // 0..511
            int row = e >> 2;
            int col = (e & 3) * 8;
            *(v8h*)(&sA[0][0] + row * 40 + col) =
                *(const v8h*)(A + (size_t)(bm + row) * K + k0 + col);
            *(v8h*)(&sB[0][0] + row * 40 + col) =
                *(const v8h*)(Bt + (size_t)(bn + row) * K + k0 + col);
        }
        if (k0 + 32 < K) {
            __builtin_prefetch(A + (size_t)(bm + (t >> 1)) * K + k0 + 32, 0, 0);
            __builtin_prefetch(Bt + (size_t)(bn + (t >> 1)) * K + k0 + 32, 0, 0);
        }
        __syncthreads();
        const f16* pa = &sA[0][0];
        const f16* pb = &sB[0][0];
#endif
        v16h af[4], bf[2];
#pragma unroll
        for (int ii = 0; ii < 4; ++ii)
            af[ii] = ldfrag(pa + (wm + ii * 16 + mr) * 40 + kb);
#pragma unroll
        for (int jj = 0; jj < 2; ++jj)
            bf[jj] = ldfrag(pb + (wn + jj * 16 + mr) * 40 + kb);
#pragma unroll
        for (int ii = 0; ii < 4; ++ii)
#pragma unroll
            for (int jj = 0; jj < 2; ++jj)
                acc[ii][jj] = __builtin_amdgcn_wmma_f32_16x16x32_f16(
                    false, af[ii], false, bf[jj], (short)0, acc[ii][jj], false, false);
        __syncthreads();
    }

    // epilogue: C layout  M = r + 8*half, N = lane&15
#pragma unroll
    for (int i = 0; i < 4; ++i)
#pragma unroll
        for (int j = 0; j < 2; ++j) {
            int ccol = bn + wn + j * 16 + mr;
            float bv = (flags & GF_BIAS) ? bias[ccol] : 0.f;
#pragma unroll
            for (int r = 0; r < 8; ++r) {
                int crow = bm + wm + i * 16 + r + 8 * half;
                float val = acc[i][j][r] + bv;
                if (flags & GF_RELU) val = fmaxf(val, 0.f);
                if (flags & GF_RES)  val += resid[(size_t)crow * N + ccol];
                if (flags & GF_F32)  outF[(size_t)crow * N + ccol] = val;
                if (flags & GF_F16)  outH[(size_t)crow * N + ccol] = (f16)val;
            }
        }
}

// -------------------- flash attention (1 wave per 16-query tile) ------------
// grid: (T/16, B*H), block: 32.  q/k/v are f16 [B,T,D] head-packed (h*64+d).
__global__ __launch_bounds__(32) void k_attn(const f16* __restrict__ q,
                                             const f16* __restrict__ kmat,
                                             const f16* __restrict__ vmat,
                                             f16* __restrict__ out) {
    __shared__ f16 sP[16 * 32];

    const int qb = blockIdx.x;
    const int bh = blockIdx.y;
    const int bb = bh / CH, hh = bh % CH;
    const int lane = threadIdx.x;
    const int half = lane >> 4;
    const int col  = lane & 15;
    const int kb   = half * 8;
    const float scale = 0.125f;          // HS^-0.5

    v16h qf[2];
#pragma unroll
    for (int c = 0; c < 2; ++c) {
        size_t base = ((size_t)(bb * CT + qb * 16 + col)) * CD + hh * CHS + c * 32 + kb;
        qf[c] = ldfrag(q + base);
    }

    v8f oacc[4];
#pragma unroll
    for (int nt = 0; nt < 4; ++nt)
#pragma unroll
        for (int r = 0; r < 8; ++r) oacc[nt][r] = 0.f;
    float mrow[8], lrow[8];
#pragma unroll
    for (int r = 0; r < 8; ++r) { mrow[r] = -1e30f; lrow[r] = 0.f; }

    const int nch = (16 * qb + 47) >> 5;
    for (int j = 0; j < nch; ++j) {
        v8f s0, s1;
#pragma unroll
        for (int r = 0; r < 8; ++r) { s0[r] = 0.f; s1[r] = 0.f; }
#pragma unroll
        for (int c = 0; c < 2; ++c) {
            size_t kb0 = ((size_t)(bb * CT + j * 32 + col))      * CD + hh * CHS + c * 32 + kb;
            size_t kb1 = ((size_t)(bb * CT + j * 32 + 16 + col)) * CD + hh * CHS + c * 32 + kb;
            v16h kf0 = ldfrag(kmat + kb0);
            v16h kf1 = ldfrag(kmat + kb1);
            s0 = __builtin_amdgcn_wmma_f32_16x16x32_f16(false, qf[c], false, kf0,
                                                        (short)0, s0, false, false);
            s1 = __builtin_amdgcn_wmma_f32_16x16x32_f16(false, qf[c], false, kf1,
                                                        (short)0, s1, false, false);
        }
        int key0 = j * 32 + col;
        int key1 = j * 32 + 16 + col;
        float p0[8], p1[8], alpha[8];
#pragma unroll
        for (int r = 0; r < 8; ++r) {
            int tq = qb * 16 + r + 8 * half;
            float a = s0[r] * scale; if (key0 > tq) a = -1e30f;
            float b = s1[r] * scale; if (key1 > tq) b = -1e30f;
            float cm = rmax16(fmaxf(a, b));
            float nm = fmaxf(mrow[r], cm);
            alpha[r] = __expf(mrow[r] - nm);
            p0[r] = __expf(a - nm);
            p1[r] = __expf(b - nm);
            lrow[r] = lrow[r] * alpha[r] + rsum16(p0[r] + p1[r]);
            mrow[r] = nm;
        }
#pragma unroll
        for (int nt = 0; nt < 4; ++nt)
#pragma unroll
            for (int r = 0; r < 8; ++r) oacc[nt][r] *= alpha[r];

        __syncthreads();
#pragma unroll
        for (int r = 0; r < 8; ++r) {
            int m = r + 8 * half;
            sP[m * 32 + col]      = (f16)p0[r];
            sP[m * 32 + 16 + col] = (f16)p1[r];
        }
        __syncthreads();
        v16h pf = ldfrag(sP + col * 32 + kb);

#pragma unroll
        for (int nt = 0; nt < 4; ++nt) {
            v16h vf;
#pragma unroll
            for (int e = 0; e < 16; ++e) {
                int klocal = (e < 8) ? (kb + e) : (kb + 8 + e);
                vf[e] = vmat[((size_t)(bb * CT + j * 32 + klocal)) * CD +
                             hh * CHS + nt * 16 + col];
            }
            oacc[nt] = __builtin_amdgcn_wmma_f32_16x16x32_f16(
                false, pf, false, vf, (short)0, oacc[nt], false, false);
        }
        __syncthreads();
    }

#pragma unroll
    for (int nt = 0; nt < 4; ++nt)
#pragma unroll
        for (int r = 0; r < 8; ++r) {
            int m = r + 8 * half;
            size_t o = ((size_t)(bb * CT + qb * 16 + m)) * CD + hh * CHS + nt * 16 + col;
            out[o] = (f16)(oacc[nt][r] / lrow[r]);
        }
}

// -------------------- loss --------------------

__global__ void k_zero1(float* p) { *p = 0.f; }

__global__ __launch_bounds__(256) void k_loss_rows(const float* __restrict__ logits,
                                                   const int* __restrict__ tgt,
                                                   float* __restrict__ acc) {
    __shared__ float red[256];
    int row = blockIdx.x, t = threadIdx.x;
    const float* lr = logits + (size_t)row * CV;
    float mx = -1e30f;
    for (int i = t; i < CV; i += 256) mx = fmaxf(mx, lr[i]);
    red[t] = mx; __syncthreads();
    for (int o = 128; o > 0; o >>= 1) { if (t < o) red[t] = fmaxf(red[t], red[t + o]); __syncthreads(); }
    float rowmax = red[0]; __syncthreads();
    float s = 0.f;
    for (int i = t; i < CV; i += 256) s += __expf(lr[i] - rowmax);
    red[t] = s; __syncthreads();
    for (int o = 128; o > 0; o >>= 1) { if (t < o) red[t] += red[t + o]; __syncthreads(); }
    if (t == 0) atomicAdd(acc, rowmax + logf(red[0]) - lr[tgt[row]]);
}

__global__ void k_loss_final(const float* acc, float* out) {
    *out = *acc * (1.f / (float)CBT);
}

// -------------------- host driver --------------------

extern "C" void kernel_launch(void* const* d_in, const int* in_sizes, int n_in,
                              void* d_out, int out_size, void* d_ws, size_t ws_size,
                              hipStream_t stream) {
    (void)in_sizes; (void)n_in; (void)out_size; (void)ws_size;
    const int*   x     = (const int*)  d_in[0];
    const int*   tgt   = (const int*)  d_in[1];
    const float* emb   = (const float*)d_in[2];
    const float* pos   = (const float*)d_in[3];
    const float* ln1_g = (const float*)d_in[4];
    const float* ln1_b = (const float*)d_in[5];
    const float* wq    = (const float*)d_in[6];
    const float* wk    = (const float*)d_in[7];
    const float* wv    = (const float*)d_in[8];
    const float* wo    = (const float*)d_in[9];
    const float* bo    = (const float*)d_in[10];
    const float* ln2_g = (const float*)d_in[11];
    const float* ln2_b = (const float*)d_in[12];
    const float* w1    = (const float*)d_in[13];
    const float* b1    = (const float*)d_in[14];
    const float* w2    = (const float*)d_in[15];
    const float* b2    = (const float*)d_in[16];
    const float* lnf_g = (const float*)d_in[17];
    const float* lnf_b = (const float*)d_in[18];
    const float* wc    = (const float*)d_in[19];
    const float* bc    = (const float*)d_in[20];

    // ---- carve workspace (~260 MB) ----
    char* p = (char*)d_ws;
    auto carve = [&](size_t bytes) -> void* {
        void* r = (void*)p;
        p += (bytes + 255) & ~(size_t)255;
        return r;
    };
    const size_t DD = (size_t)CD * CD;
    f16*   w16   = (f16*)  carve(sizeof(f16) * CL * 12 * DD);   // transposed weights
    f16*   wc16  = (f16*)  carve(sizeof(f16) * (size_t)CD * CV); // wc^T [V][D]
    float* h     = (float*)carve(sizeof(float) * (size_t)CBT * CD);
    f16*   a16   = (f16*)  carve(sizeof(f16) * (size_t)CBT * CD);
    f16*   q16   = (f16*)  carve(sizeof(f16) * (size_t)CBT * CD);
    f16*   k16   = (f16*)  carve(sizeof(f16) * (size_t)CBT * CD);
    f16*   v16   = (f16*)  carve(sizeof(f16) * (size_t)CBT * CD);
    f16*   at16  = (f16*)  carve(sizeof(f16) * (size_t)CBT * CD);
    f16*   mh16  = (f16*)  carve(sizeof(f16) * (size_t)CBT * CFD);
    f16*   hf16  = (f16*)  carve(sizeof(f16) * (size_t)CBT * CD);
    float* lacc  = (float*)carve(256);

    // fp32 [K][N] -> f16 transposed [N][K]
    auto cvt_tr = [&](const float* src, f16* dst, int Kd, int Nd) {
        k_cvt_tr<<<dim3(Nd / 32, Kd / 32), 256, 0, stream>>>(src, dst, Kd, Nd);
    };

    // ---- convert + transpose weights to f16 (every call; deterministic) ----
    for (int l = 0; l < CL; ++l) {
        f16* base = w16 + (size_t)l * 12 * DD;
        cvt_tr(wq + (size_t)l * DD,     base + 0 * DD, CD, CD);
        cvt_tr(wk + (size_t)l * DD,     base + 1 * DD, CD, CD);
        cvt_tr(wv + (size_t)l * DD,     base + 2 * DD, CD, CD);
        cvt_tr(wo + (size_t)l * DD,     base + 3 * DD, CD, CD);
        cvt_tr(w1 + (size_t)l * 4 * DD, base + 4 * DD, CD, CFD);   // -> [4D][D]
        cvt_tr(w2 + (size_t)l * 4 * DD, base + 8 * DD, CFD, CD);   // -> [D][4D]
    }
    cvt_tr(wc, wc16, CD, CV);                                      // -> [V][D]

    // ---- embeddings ----
    k_embed<<<dim3(CBT), 256, 0, stream>>>(x, emb, pos, h);

    dim3 blk(256);
    dim3 gD (CD  / 128, CBT / 128);
    dim3 gFD(CFD / 128, CBT / 128);
    dim3 gV (CV  / 128, CBT / 128);

    for (int l = 0; l < CL; ++l) {
        f16* base = w16 + (size_t)l * 12 * DD;
        k_layernorm<<<dim3(CBT), 256, 0, stream>>>(h, ln1_g + l * CD, ln1_b + l * CD, a16);
        k_gemm_wmma<<<gD, blk, 0, stream>>>(a16, base + 0 * DD, nullptr, nullptr,
                                            nullptr, q16, CBT, CD, CD, GF_F16);
        k_gemm_wmma<<<gD, blk, 0, stream>>>(a16, base + 1 * DD, nullptr, nullptr,
                                            nullptr, k16, CBT, CD, CD, GF_F16);
        k_gemm_wmma<<<gD, blk, 0, stream>>>(a16, base + 2 * DD, nullptr, nullptr,
                                            nullptr, v16, CBT, CD, CD, GF_F16);
        k_attn<<<dim3(CT / 16, CB * CH), 32, 0, stream>>>(q16, k16, v16, at16);
        k_gemm_wmma<<<gD, blk, 0, stream>>>(at16, base + 3 * DD, bo + l * CD, h,
                                            h, nullptr, CBT, CD, CD,
                                            GF_BIAS | GF_RES | GF_F32);
        k_layernorm<<<dim3(CBT), 256, 0, stream>>>(h, ln2_g + l * CD, ln2_b + l * CD, a16);
        k_gemm_wmma<<<gFD, blk, 0, stream>>>(a16, base + 4 * DD, b1 + (size_t)l * CFD,
                                             nullptr, nullptr, mh16, CBT, CFD, CD,
                                             GF_BIAS | GF_RELU | GF_F16);
        k_gemm_wmma<<<gD, blk, 0, stream>>>(mh16, base + 8 * DD, b2 + l * CD, h,
                                            h, nullptr, CBT, CD, CFD,
                                            GF_BIAS | GF_RES | GF_F32);
    }

    k_layernorm<<<dim3(CBT), 256, 0, stream>>>(h, lnf_g, lnf_b, hf16);
    float* logits = (float*)d_out;
    k_gemm_wmma<<<gV, blk, 0, stream>>>(hf16, wc16, bc, nullptr,
                                        logits, nullptr, CBT, CV, CD,
                                        GF_BIAS | GF_F32);
    k_zero1<<<1, 1, 0, stream>>>(lacc);
    k_loss_rows<<<dim3(CBT), 256, 0, stream>>>(logits, tgt, lacc);
    k_loss_final<<<1, 1, 0, stream>>>(lacc, logits + (size_t)CBT * CV);
}
